// SAIN_39779987096137
// MI455X (gfx1250) — matrix operand order repaired and checked by reading, not compiled, not run on hardware
//
#include <hip/hip_runtime.h>
#include <stdint.h>

typedef __attribute__((ext_vector_type(16))) __bf16    v16bf;
typedef __attribute__((ext_vector_type(2)))  __bf16    bf16x2;
typedef __attribute__((ext_vector_type(8)))  float     v8f;
typedef __attribute__((ext_vector_type(8)))  uint32_t  v8u;

#define B_DIM   2
#define N_NODES 20000
#define E_EDGES 640000

#define EW   4        // waves per block
#define TPB  (EW*32)

struct MLPp { const float* w[5]; const float* b[5]; };

// ---------------------------------------------------------------------------
__global__ void zero_kernel(float* p, int n) {
  for (int i = blockIdx.x * blockDim.x + threadIdx.x; i < n; i += gridDim.x * blockDim.x)
    p[i] = 0.f;
}

// ---------------------------------------------------------------------------
__device__ inline v8f wmma_bf16(v16bf a, v16bf b, v8f c) {
  return __builtin_amdgcn_wmma_f32_16x16x32_bf16(false, a, false, b, (short)0, c,
                                                 false, false);
}

__device__ inline uint32_t swapx16(uint32_t x) {
  // ds_swizzle SWAPX16: group-of-32, xor_mask=0x10, and_mask=0x1f
  return (uint32_t)__builtin_amdgcn_ds_swizzle((int)x, 0x401F);
}

// Packed bf16 ReLU: max_num against +0 on both halves (VOP3P).
__device__ inline uint32_t relu_pk_bf16(uint32_t x) {
  uint32_t r;
  asm("v_pk_max_num_bf16 %0, %1, 0" : "=v"(r) : "v"(x));
  return r;
}

// Pack two floats into one dword of bf16 pairs (-> v_cvt_pk_bf16_f32).
__device__ inline uint32_t pack2(float a, float b) {
  bf16x2 h;
  h[0] = (__bf16)a;
  h[1] = (__bf16)b;
  return __builtin_bit_cast(uint32_t, h);
}

// Pack one D tile into 4 dwords of bf16 pairs, then packed-bf16 ReLU.
__device__ inline void packP(v8f c, uint32_t* P) {
  #pragma unroll
  for (int i = 0; i < 4; ++i)
    P[i] = relu_pk_bf16(pack2(c[2 * i], c[2 * i + 1]));
}

__device__ inline v16bf to_v16bf(const uint32_t* u) {
  v8u t;
  #pragma unroll
  for (int i = 0; i < 8; ++i) t[i] = u[i];
  return __builtin_bit_cast(v16bf, t);
}

// A fragment = W^T tile: element (m = t16+ln, k) = w[k*Nout + m].
// A layout used here (contiguous k, not the ISA table order): lane = M + 16*g ; slot j -> k = g*16+j.
// Optional bias row at k == kReal (value bias[m]); rows m >= mReal are zero.
__device__ inline v16bf load_wfrag(const float* w, const float* bias, int Nout,
                                   int mReal, int kReal, int t16, int pk32,
                                   bool addBias, int ln, int g) {
  v16bf a;
  const int m = t16 + ln;
  #pragma unroll
  for (int j = 0; j < 16; ++j) {
    const int k = pk32 + g * 16 + j;
    float v = 0.f;
    if (m < mReal) {
      if (k < kReal) v = w[k * Nout + m];
      else if (addBias && k == kReal) v = bias[m];
    }
    a[j] = (__bf16)v;
  }
  return a;
}

// Bias-row A fragment: only element (m, k=0) = bias[m] nonzero (slot0, g=0).
__device__ inline v16bf load_bias_afrag(const float* bias, int mReal, int t16,
                                        int ln, int g) {
  v16bf a = {};
  const int m = t16 + ln;
  float v = 0.f;
  if (g == 0 && m < mReal) v = bias[m];
  a[0] = (__bf16)v;
  return a;
}

// Constant B fragment: slot0 = 1.0 on g=0 lanes (k=0 row of ones).
__device__ inline v16bf ones_bfrag(int g) {
  v16bf b = {};
  b[0] = (g == 0) ? (__bf16)1.0f : (__bf16)0.0f;
  return b;
}

// Build B fragment (32 k x 16 cols) from two packed D tiles.
// T0 = tile 2pk (k lo 16), T1 = tile 2pk+1 (k hi 16).
// B layout used here (contiguous k, as in A): lane = col + 16*(k/16); slot j -> k = g*16 + j.
__device__ inline v16bf buildB2(const uint32_t* T0, const uint32_t* T1, int g) {
  uint32_t out[8];
  #pragma unroll
  for (int i = 0; i < 4; ++i) {
    const uint32_t Z = g ? T1[i] : T0[i];   // own copy of my tile
    const uint32_t Y = g ? T0[i] : T1[i];   // own copy of partner's tile
    const uint32_t S = swapx16(Y);          // partner's copy of MY tile
    out[i]     = g ? S : Z;                 // slots 0..7  (features 0..7)
    out[4 + i] = g ? Z : S;                 // slots 8..15 (features 8..15)
  }
  return to_v16bf(out);
}

// Last-layer B (K=16 real + bias row at k=16): g=0 half from tile T0,
// g=1 half = {1.0bf16 at slot 0} to activate the bias row of A.
__device__ inline v16bf buildB_last16(const uint32_t* T0, int g) {
  uint32_t out[8];
  #pragma unroll
  for (int i = 0; i < 4; ++i) {
    const uint32_t S = swapx16(T0[i]);      // partner's copy of tile 0
    out[i]     = g ? 0u : T0[i];
    out[4 + i] = g ? 0u : S;
  }
  if (g) out[0] = 0x3F80u;                  // bf16(1.0) in slot 0
  return to_v16bf(out);
}

// ---------------------------------------------------------------------------
// Edge MLP 16->128->64->32->16->16, weights VGPR-resident, + scatter-add.
__global__ __launch_bounds__(TPB)
void edge_kernel(const float* __restrict__ rel, const int* __restrict__ recv,
                 float* __restrict__ agg, MLPp p) {
  const int lane = threadIdx.x & 31, wave = threadIdx.x >> 5;
  const int ln = lane & 15, g = lane >> 4;

  // ---- preload all weight fragments into VGPRs (once per wave) ----
  v16bf A0[8], A1[4][4], A2[2][2], A3, A4;
  v16bf A1b[4], A2b[2], A3b;
  #pragma unroll
  for (int t = 0; t < 8; ++t)
    A0[t] = load_wfrag(p.w[0], p.b[0], 128, 128, 16, t * 16, 0, true, ln, g);
  #pragma unroll
  for (int t = 0; t < 4; ++t) {
    #pragma unroll
    for (int pk = 0; pk < 4; ++pk)
      A1[t][pk] = load_wfrag(p.w[1], p.b[1], 64, 64, 128, t * 16, pk * 32, false, ln, g);
    A1b[t] = load_bias_afrag(p.b[1], 64, t * 16, ln, g);
  }
  #pragma unroll
  for (int t = 0; t < 2; ++t) {
    #pragma unroll
    for (int pk = 0; pk < 2; ++pk)
      A2[t][pk] = load_wfrag(p.w[2], p.b[2], 32, 32, 64, t * 16, pk * 32, false, ln, g);
    A2b[t] = load_bias_afrag(p.b[2], 32, t * 16, ln, g);
  }
  A3  = load_wfrag(p.w[3], p.b[3], 16, 16, 32, 0, 0, false, ln, g);
  A3b = load_bias_afrag(p.b[3], 16, 0, ln, g);
  A4  = load_wfrag(p.w[4], p.b[4], 16, 16, 16, 0, 0, true, ln, g);
  const v16bf bONE = ones_bfrag(g);

  const int tilesPerBatch = E_EDGES / 16;
  const int ntile = B_DIM * tilesPerBatch;
  for (int tile = blockIdx.x * EW + wave; tile < ntile; tile += gridDim.x * EW) {
    const int bb = tile / tilesPerBatch;
    const int e0 = (tile % tilesPerBatch) * 16;

    // B0: g=0 lanes carry the 16 features of edge e0+ln (k=0..15);
    //     g=1 lanes: slot0 = 1.0 to activate bias row k=16 of A0.
    uint32_t d[8];
    if (g == 0) {
      const float4* rp = (const float4*)(rel + ((size_t)bb * E_EDGES + e0 + ln) * 16);
      #pragma unroll
      for (int q = 0; q < 4; ++q) {
        const float4 f = rp[q];
        d[2 * q]     = pack2(f.x, f.y);
        d[2 * q + 1] = pack2(f.z, f.w);
      }
    } else {
      #pragma unroll
      for (int i = 0; i < 8; ++i) d[i] = 0u;
      d[0] = 0x3F80u;
    }
    const v16bf b0 = to_v16bf(d);

    // L0: 16->128 (bias folded into A0 k=16 row)
    uint32_t P[8][4];
    #pragma unroll
    for (int t = 0; t < 8; ++t) {
      v8f c = {};
      c = wmma_bf16(A0[t], b0, c);
      packP(c, P[t]);
    }
    // L1: 128->64, bias via bias-row WMMA (c starts at inline 0)
    v8f a1[4];
    #pragma unroll
    for (int t = 0; t < 4; ++t) { v8f z = {}; a1[t] = wmma_bf16(A1b[t], bONE, z); }
    #pragma unroll
    for (int pk = 0; pk < 4; ++pk) {
      const v16bf bf = buildB2(P[2 * pk], P[2 * pk + 1], g);
      #pragma unroll
      for (int t = 0; t < 4; ++t) a1[t] = wmma_bf16(A1[t][pk], bf, a1[t]);
    }
    uint32_t Q[4][4];
    #pragma unroll
    for (int t = 0; t < 4; ++t) packP(a1[t], Q[t]);
    // L2: 64->32
    v8f a2[2];
    #pragma unroll
    for (int t = 0; t < 2; ++t) { v8f z = {}; a2[t] = wmma_bf16(A2b[t], bONE, z); }
    #pragma unroll
    for (int pk = 0; pk < 2; ++pk) {
      const v16bf bf = buildB2(Q[2 * pk], Q[2 * pk + 1], g);
      #pragma unroll
      for (int t = 0; t < 2; ++t) a2[t] = wmma_bf16(A2[t][pk], bf, a2[t]);
    }
    uint32_t R[2][4];
    #pragma unroll
    for (int t = 0; t < 2; ++t) packP(a2[t], R[t]);
    // L3: 32->16
    v8f a3;
    { v8f z = {}; a3 = wmma_bf16(A3b, bONE, z); }
    a3 = wmma_bf16(A3, buildB2(R[0], R[1], g), a3);
    uint32_t S[4];
    packP(a3, S);
    // L4: 16->16 linear (bias row k=16 in A4; no ReLU)
    v8f msg = {};
    msg = wmma_bf16(A4, buildB_last16(S, g), msg);

    // scatter: lane holds features 8*g..8*g+7 of edge e0+ln
    const int dst = recv[e0 + ln];
    float* basep = agg + ((size_t)bb * N_NODES + dst) * 16 + 8 * g;
    #pragma unroll
    for (int r = 0; r < 8; ++r) atomicAdd(basep + r, msg[r]);
  }
}

// ---------------------------------------------------------------------------
// Node MLP concat(dyn14, agg16) -> 30->128->64->32->16->6.
__global__ __launch_bounds__(TPB)
void node_kernel(const float* __restrict__ dyn, const float* __restrict__ agg,
                 float* __restrict__ out, MLPp p) {
  const int lane = threadIdx.x & 31, wave = threadIdx.x >> 5;
  const int ln = lane & 15, g = lane >> 4;

  v16bf A0[8], A1[4][4], A2[2][2], A3, A4;
  v16bf A1b[4], A2b[2], A3b;
  #pragma unroll
  for (int t = 0; t < 8; ++t)
    A0[t] = load_wfrag(p.w[0], p.b[0], 128, 128, 30, t * 16, 0, true, ln, g);
  #pragma unroll
  for (int t = 0; t < 4; ++t) {
    #pragma unroll
    for (int pk = 0; pk < 4; ++pk)
      A1[t][pk] = load_wfrag(p.w[1], p.b[1], 64, 64, 128, t * 16, pk * 32, false, ln, g);
    A1b[t] = load_bias_afrag(p.b[1], 64, t * 16, ln, g);
  }
  #pragma unroll
  for (int t = 0; t < 2; ++t) {
    #pragma unroll
    for (int pk = 0; pk < 2; ++pk)
      A2[t][pk] = load_wfrag(p.w[2], p.b[2], 32, 32, 64, t * 16, pk * 32, false, ln, g);
    A2b[t] = load_bias_afrag(p.b[2], 32, t * 16, ln, g);
  }
  A3  = load_wfrag(p.w[3], p.b[3], 16, 16, 32, 0, 0, false, ln, g);
  A3b = load_bias_afrag(p.b[3], 16, 0, ln, g);
  A4  = load_wfrag(p.w[4], p.b[4], 6, 6, 16, 0, 0, true, ln, g);   // m >= 6 zero
  const v16bf bONE = ones_bfrag(g);

  const int tilesPerBatch = N_NODES / 16;
  const int ntile = B_DIM * tilesPerBatch;
  for (int tile = blockIdx.x * EW + wave; tile < ntile; tile += gridDim.x * EW) {
    const int bb = tile / tilesPerBatch;
    const int n0 = (tile % tilesPerBatch) * 16;
    const int node = n0 + ln;

    // B0: lane (ln,g) slot j -> k = g*16+j: k<14 dyn, 14<=k<30 agg, k==30 bias 1.0
    const float* dp = dyn + ((size_t)bb * N_NODES + node) * 14;
    const float* ap = agg + ((size_t)bb * N_NODES + node) * 16;
    v16bf b0;
    #pragma unroll
    for (int j = 0; j < 16; ++j) {
      const int k = g * 16 + j;
      float v = 0.f;
      if (k < 14)       v = dp[k];
      else if (k < 30)  v = ap[k - 14];
      else if (k == 30) v = 1.f;
      b0[j] = (__bf16)v;
    }

    uint32_t P[8][4];
    #pragma unroll
    for (int t = 0; t < 8; ++t) {
      v8f c = {};
      c = wmma_bf16(A0[t], b0, c);
      packP(c, P[t]);
    }
    v8f a1[4];
    #pragma unroll
    for (int t = 0; t < 4; ++t) { v8f z = {}; a1[t] = wmma_bf16(A1b[t], bONE, z); }
    #pragma unroll
    for (int pk = 0; pk < 4; ++pk) {
      const v16bf bf = buildB2(P[2 * pk], P[2 * pk + 1], g);
      #pragma unroll
      for (int t = 0; t < 4; ++t) a1[t] = wmma_bf16(A1[t][pk], bf, a1[t]);
    }
    uint32_t Q[4][4];
    #pragma unroll
    for (int t = 0; t < 4; ++t) packP(a1[t], Q[t]);
    v8f a2[2];
    #pragma unroll
    for (int t = 0; t < 2; ++t) { v8f z = {}; a2[t] = wmma_bf16(A2b[t], bONE, z); }
    #pragma unroll
    for (int pk = 0; pk < 2; ++pk) {
      const v16bf bf = buildB2(Q[2 * pk], Q[2 * pk + 1], g);
      #pragma unroll
      for (int t = 0; t < 2; ++t) a2[t] = wmma_bf16(A2[t][pk], bf, a2[t]);
    }
    uint32_t R[2][4];
    #pragma unroll
    for (int t = 0; t < 2; ++t) packP(a2[t], R[t]);
    v8f a3;
    { v8f z = {}; a3 = wmma_bf16(A3b, bONE, z); }
    a3 = wmma_bf16(A3, buildB2(R[0], R[1], g), a3);
    uint32_t S[4];
    packP(a3, S);
    v8f o = {};
    o = wmma_bf16(A4, buildB_last16(S, g), o);

    // lane (ln, g=0) holds out-features 0..7 of node n0+ln; only f<6 valid
    if (g == 0) {
      float* op = out + ((size_t)bb * N_NODES + node) * 6;
      #pragma unroll
      for (int r = 0; r < 6; ++r) op[r] = o[r];
    }
  }
}

// ---------------------------------------------------------------------------
extern "C" void kernel_launch(void* const* d_in, const int* in_sizes, int n_in,
                              void* d_out, int out_size, void* d_ws, size_t ws_size,
                              hipStream_t stream) {
  const float* dyn  = (const float*)d_in[0];
  const float* rel  = (const float*)d_in[1];
  const int*   recv = (const int*)d_in[3];          // send (d_in[2]) unused by reference

  MLPp frel, fdyn;
  for (int i = 0; i < 5; ++i) {
    frel.w[i] = (const float*)d_in[4  + 2 * i];
    frel.b[i] = (const float*)d_in[5  + 2 * i];
    fdyn.w[i] = (const float*)d_in[14 + 2 * i];
    fdyn.b[i] = (const float*)d_in[15 + 2 * i];
  }

  float* agg = (float*)d_ws;                         // [B][N][16] f32
  float* out = (float*)d_out;                        // [B][N][6]  f32

  zero_kernel<<<256, 256, 0, stream>>>(agg, B_DIM * N_NODES * 16);
  edge_kernel<<<640, TPB, 0, stream>>>(rel, recv, agg, frel);
  node_kernel<<<640, TPB, 0, stream>>>(dyn, agg, out, fdyn);
}
